// MorphoGradDGNN_52192442581529
// MI455X (gfx1250) — matrix operand order repaired and checked by reading, not compiled, not run on hardware
//
#include <hip/hip_runtime.h>
#include <hip/hip_bf16.h>

typedef __bf16 bf16_t;
typedef __bf16 v8bf  __attribute__((ext_vector_type(8)));
typedef __bf16 v16bf __attribute__((ext_vector_type(16)));
typedef float  v8f   __attribute__((ext_vector_type(8)));

#define NPTS  16384
#define KNN   20
#define NCLS  40
#define CFEAT 64
#define FINF  3.0e38f

// ---------------------------------------------------------------------------
// WMMA fragment loaders (bf16 16x16x32, wave32, contiguous k = 16*(lane/16) + s in A and B)
// ---------------------------------------------------------------------------

// A fragment 16x32 bf16 from row-major A[lda]: lane<16 -> M=lane, K=0..15;
// lane>=16 -> M=lane-16, K=16..31. Two contiguous 16B loads per lane.
__device__ inline v16bf load_a_frag(const bf16_t* A, int lda, int m0, int k0, int lane) {
    const bf16_t* base = A + (size_t)(m0 + (lane & 15)) * lda + k0 + ((lane < 16) ? 0 : 16);
    v8bf lo = *(const v8bf*)(base);
    v8bf hi = *(const v8bf*)(base + 8);
    return __builtin_shufflevector(lo, hi, 0,1,2,3,4,5,6,7,8,9,10,11,12,13,14,15);
}

// B fragment 32x16 bf16 for Gram matrix (B = P^T): lane<16 -> N=lane, K=0..15;
// lane>=16 -> N=lane-16, K=16..31.  Column n of P^T == row (j0+n) of P, so this
// is one contiguous 32B load per lane from row-major P (ld = 64).
__device__ inline v16bf load_b_gram(const bf16_t* P, int j0, int k0, int lane) {
    const bf16_t* base = P + (size_t)(j0 + (lane & 15)) * CFEAT + k0 + ((lane < 16) ? 0 : 16);
    return *(const v16bf*)base;
}

// ---------------------------------------------------------------------------
// Elementwise helpers
// ---------------------------------------------------------------------------

// f32 -> bf16 with optional zero-padding of trailing channels
__global__ void f32_to_bf16_pad(const float* __restrict__ src, bf16_t* __restrict__ dst,
                                int n, int cin, int cpad) {
    int t = blockIdx.x * blockDim.x + threadIdx.x;
    if (t >= n * cpad) return;
    int r = t / cpad, c = t % cpad;
    float v = (c < cin) ? src[(size_t)r * cin + c] : 0.0f;
    dst[t] = (bf16_t)v;
}

// Pre-swizzle row-major W[ksrc x ncol] (zero-padded to kpad) into B-fragment-major
// bf16: frag(kt,nt) -> 32 lanes x 16 contiguous bf16 each.
__global__ void swizzle_wB(const float* __restrict__ W, bf16_t* __restrict__ out,
                           int ksrc, int kpad, int ncol) {
    int t = blockIdx.x * blockDim.x + threadIdx.x;
    if (t >= kpad * ncol) return;
    int e    = t & 15;
    int l    = (t >> 4) & 31;
    int frag = t >> 9;
    int ntiles = ncol >> 4;
    int nt = frag % ntiles;
    int kt = frag / ntiles;
    int k = kt * 32 + ((l < 16) ? 0 : 16) + e;
    int n = nt * 16 + (l & 15);
    float v = (k < ksrc) ? W[(size_t)k * ncol + n] : 0.0f;
    out[t] = (bf16_t)v;
}

__global__ void row_sqnorm(const bf16_t* __restrict__ P, float* __restrict__ sq, int n) {
    int r = blockIdx.x * blockDim.x + threadIdx.x;
    if (r >= n) return;
    float s = 0.0f;
    #pragma unroll 8
    for (int c = 0; c < CFEAT; ++c) {
        float v = (float)P[(size_t)r * CFEAT + c];
        s += v * v;
    }
    sq[r] = s;
}

// ---------------------------------------------------------------------------
// Generic WMMA GEMM:  C[M x Ncol] = relu?( A[M x K](bf16) * Bfrag + bias )
// One wave per 16x16 C tile, 4 waves / block.
// ---------------------------------------------------------------------------
__global__ __launch_bounds__(128) void gemm_bf16_wmma(
        const bf16_t* __restrict__ A, const bf16_t* __restrict__ Bfrag,
        const float* __restrict__ bias, float* __restrict__ C,
        int M, int Kdim, int Ncol, int relu) {
    int lane  = threadIdx.x & 31;
    int wave  = threadIdx.x >> 5;
    int mtile = blockIdx.x;
    int ntile = blockIdx.y * 4 + wave;
    int ntiles = Ncol >> 4;
    int m0 = mtile * 16, n0 = ntile * 16;

    v8f c = {};
    int kts = Kdim >> 5;
    for (int kt = 0; kt < kts; ++kt) {
        v16bf a = load_a_frag(A, Kdim, m0, kt * 32, lane);
        v16bf b = *(const v16bf*)(Bfrag + ((((size_t)kt * ntiles + ntile) * 32 + lane) << 4));
        c = __builtin_amdgcn_wmma_f32_16x16x32_bf16(false, a, false, b, (short)0, c, false, false);
    }

    int n     = n0 + (lane & 15);
    int mbase = m0 + ((lane < 16) ? 0 : 8);
    float bv  = bias[n];
    #pragma unroll
    for (int r = 0; r < 8; ++r) {
        float v = c[r] + bv;
        if (relu) v = fmaxf(v, 0.0f);
        C[(size_t)(mbase + r) * Ncol + n] = v;
    }
}

// ---------------------------------------------------------------------------
// Fused WMMA Gram-tile + top-K(=20) kNN. One block (8 waves) per 16-row tile.
// Each wave sweeps 1/8 of the column tiles; per-row top-k lists kept in LDS.
// ---------------------------------------------------------------------------
__global__ __launch_bounds__(256) void knn_wmma(
        const bf16_t* __restrict__ P, const float* __restrict__ sq,
        int* __restrict__ knn_out, int n) {
    __shared__ float s_dist[8][16][16];
    __shared__ float s_topv[8][16][KNN];
    __shared__ int   s_topi[8][16][KNN];

    int lane = threadIdx.x & 31;
    int wave = threadIdx.x >> 5;
    int i0   = blockIdx.x * 16;

    // A fragments for this row tile (K = 64 -> two fragments), loaded once.
    v16bf a0 = load_a_frag(P, CFEAT, i0, 0,  lane);
    v16bf a1 = load_a_frag(P, CFEAT, i0, 32, lane);

    for (int t = lane; t < 16 * KNN; t += 32) {
        s_topv[wave][t / KNN][t % KNN] = FINF;
        s_topi[wave][t / KNN][t % KNN] = 0;
    }
    asm volatile("s_wait_dscnt 0x0" ::: "memory");

    float worstVal = FINF;
    int   worstSlot = 0;
    int   r   = lane & 15;
    float sqi = sq[i0 + r];

    int jts = n >> 4;
    for (int jt = wave; jt < jts; jt += 8) {
        int j0 = jt * 16;
        v16bf b0 = load_b_gram(P, j0, 0,  lane);
        v16bf b1 = load_b_gram(P, j0, 32, lane);
        v8f c = {};
        c = __builtin_amdgcn_wmma_f32_16x16x32_bf16(false, a0, false, b0, (short)0, c, false, false);
        c = __builtin_amdgcn_wmma_f32_16x16x32_bf16(false, a1, false, b1, (short)0, c, false, false);

        // C layout: lane<16 -> N=lane, M=0..7; lane>=16 -> N=lane-16, M=8..15
        int   ncol  = lane & 15;
        int   mbase = (lane < 16) ? 0 : 8;
        float sqj   = sq[j0 + ncol];
        #pragma unroll
        for (int rr = 0; rr < 8; ++rr) {
            int gi = i0 + mbase + rr, gj = j0 + ncol;
            float d = (gi == gj) ? FINF : (sqj - 2.0f * c[rr]);   // + sq[i] added at select
            s_dist[wave][mbase + rr][ncol] = d;
        }
        asm volatile("s_wait_dscnt 0x0" ::: "memory");

        if (lane < 16) {
            #pragma unroll 1
            for (int cc = 0; cc < 16; ++cc) {
                float d = sqi + s_dist[wave][r][cc];
                if (d < worstVal) {
                    s_topv[wave][r][worstSlot] = d;
                    s_topi[wave][r][worstSlot] = j0 + cc;
                    float wv = -FINF; int wsl = 0;
                    #pragma unroll
                    for (int t = 0; t < KNN; ++t) {
                        float v = s_topv[wave][r][t];
                        if (v > wv) { wv = v; wsl = t; }
                    }
                    worstVal = wv; worstSlot = wsl;
                }
            }
        }
        asm volatile("s_wait_dscnt 0x0" ::: "memory");
    }

    __syncthreads();

    // Merge 8 per-wave lists into wave-0's list; rows handled by threads 0..15.
    if (threadIdx.x < 16) {
        int rr = threadIdx.x;
        float wv = -FINF; int wsl = 0;
        for (int t = 0; t < KNN; ++t) {
            float v = s_topv[0][rr][t];
            if (v > wv) { wv = v; wsl = t; }
        }
        for (int w = 1; w < 8; ++w) {
            for (int t = 0; t < KNN; ++t) {
                float d = s_topv[w][rr][t];
                if (d < wv) {
                    s_topv[0][rr][wsl] = d;
                    s_topi[0][rr][wsl] = s_topi[w][rr][t];
                    wv = -FINF; wsl = 0;
                    for (int u = 0; u < KNN; ++u) {
                        float v = s_topv[0][rr][u];
                        if (v > wv) { wv = v; wsl = u; }
                    }
                }
            }
        }
        for (int t = 0; t < KNN; ++t)
            knn_out[(size_t)(i0 + rr) * KNN + t] = s_topi[0][rr][t];
    }
}

// ---------------------------------------------------------------------------
// Morphological gather: out = max_k dil[nbr] - min_k ero[nbr]
// writes next-stage bf16 features + f32 concat slice.
// ---------------------------------------------------------------------------
__global__ void gather_morpho(const float* __restrict__ hd, const float* __restrict__ he,
                              const int* __restrict__ knn_idx,
                              bf16_t* __restrict__ pbf, float* __restrict__ cat,
                              int cat_ld, int coff, int n) {
    int t = blockIdx.x * blockDim.x + threadIdx.x;
    if (t >= n * CFEAT) return;
    int node = t / CFEAT, c = t % CFEAT;
    const int* kb = knn_idx + (size_t)node * KNN;
    float mx = -FINF, mn = FINF;
    #pragma unroll 4
    for (int k = 0; k < KNN; ++k) {
        int nb = kb[k];
        mx = fmaxf(mx, hd[(size_t)nb * CFEAT + c]);
        mn = fminf(mn, he[(size_t)nb * CFEAT + c]);
    }
    float v = mx - mn;
    pbf[t] = (bf16_t)v;
    cat[(size_t)node * cat_ld + coff + c] = v;
}

// ---------------------------------------------------------------------------
// Final layer (128 -> 40) + log_softmax. One block (64 threads) per row.
// ---------------------------------------------------------------------------
__global__ __launch_bounds__(64) void head_logsoftmax(
        const float* __restrict__ h3, const float* __restrict__ Wout,
        const float* __restrict__ bout, float* __restrict__ out, int n) {
    __shared__ float sh[128];
    __shared__ float sl[NCLS];
    int row = blockIdx.x;
    int t = threadIdx.x;
    sh[t]      = h3[(size_t)row * 128 + t];
    sh[t + 64] = h3[(size_t)row * 128 + 64 + t];
    __syncthreads();
    if (t < NCLS) {
        float acc = bout[t];
        #pragma unroll 8
        for (int k = 0; k < 128; ++k) acc += sh[k] * Wout[(size_t)k * NCLS + t];
        sl[t] = acc;
    }
    __syncthreads();
    if (t < NCLS) {
        float m = -FINF;
        for (int k = 0; k < NCLS; ++k) m = fmaxf(m, sl[k]);
        float s = 0.0f;
        for (int k = 0; k < NCLS; ++k) s += __expf(sl[k] - m);
        out[(size_t)row * NCLS + t] = sl[t] - m - __logf(s);
    }
}

// ---------------------------------------------------------------------------
// Host-side launch
// ---------------------------------------------------------------------------
extern "C" void kernel_launch(void* const* d_in, const int* in_sizes, int n_in,
                              void* d_out, int out_size, void* d_ws, size_t ws_size,
                              hipStream_t stream) {
    const float* x     = (const float*)d_in[0];
    const float* Wd1   = (const float*)d_in[1];  const float* bd1   = (const float*)d_in[2];
    const float* We1   = (const float*)d_in[3];  const float* be1   = (const float*)d_in[4];
    const float* Wd2   = (const float*)d_in[5];  const float* bd2   = (const float*)d_in[6];
    const float* We2   = (const float*)d_in[7];  const float* be2   = (const float*)d_in[8];
    const float* Wlin1 = (const float*)d_in[9];  const float* blin1 = (const float*)d_in[10];
    const float* Wm1   = (const float*)d_in[11]; const float* bm1   = (const float*)d_in[12];
    const float* Wm2   = (const float*)d_in[13]; const float* bm2   = (const float*)d_in[14];
    const float* Wout  = (const float*)d_in[15]; const float* bout  = (const float*)d_in[16];

    const int N = NPTS;
    char* ws = (char*)d_ws;
    size_t off = 0;
    auto take = [&](size_t bytes) -> char* {
        char* p = ws + off;
        off = (off + bytes + 255) & ~(size_t)255;
        return p;
    };

    bf16_t* pbf  = (bf16_t*)take((size_t)N * 64 * 2);
    float*  sq   = (float*) take((size_t)N * 4);
    float*  hd   = (float*) take((size_t)N * 64 * 4);
    float*  he   = (float*) take((size_t)N * 64 * 4);
    int*    knn  = (int*)   take((size_t)N * KNN * 4);
    float*  cat  = (float*) take((size_t)N * 192 * 4);
    bf16_t* catb = (bf16_t*)take((size_t)N * 192 * 2);
    float*  h1   = (float*) take((size_t)N * 1024 * 4);
    bf16_t* h1b  = (bf16_t*)take((size_t)N * 1024 * 2);
    float*  h2   = (float*) take((size_t)N * 256 * 4);
    bf16_t* h2b  = (bf16_t*)take((size_t)N * 256 * 2);
    float*  h3   = (float*) take((size_t)N * 128 * 4);
    bf16_t* wd1s = (bf16_t*)take(64 * 64 * 2);
    bf16_t* we1s = (bf16_t*)take(64 * 64 * 2);
    bf16_t* wd2s = (bf16_t*)take(64 * 64 * 2);
    bf16_t* we2s = (bf16_t*)take(64 * 64 * 2);
    bf16_t* wl1s = (bf16_t*)take((size_t)192 * 1024 * 2);
    bf16_t* wm1s = (bf16_t*)take((size_t)1024 * 256 * 2);
    bf16_t* wm2s = (bf16_t*)take((size_t)256 * 128 * 2);

    // --- weight pre-swizzle (B-fragment-major bf16, K zero-padded) ---
    swizzle_wB<<<(64 * 64 + 255) / 256, 256, 0, stream>>>(Wd1, wd1s, 3, 64, 64);
    swizzle_wB<<<(64 * 64 + 255) / 256, 256, 0, stream>>>(We1, we1s, 3, 64, 64);
    swizzle_wB<<<(64 * 64 + 255) / 256, 256, 0, stream>>>(Wd2, wd2s, 64, 64, 64);
    swizzle_wB<<<(64 * 64 + 255) / 256, 256, 0, stream>>>(We2, we2s, 64, 64, 64);
    swizzle_wB<<<(192 * 1024 + 255) / 256, 256, 0, stream>>>(Wlin1, wl1s, 192, 192, 1024);
    swizzle_wB<<<(1024 * 256 + 255) / 256, 256, 0, stream>>>(Wm1, wm1s, 1024, 1024, 256);
    swizzle_wB<<<(256 * 128 + 255) / 256, 256, 0, stream>>>(Wm2, wm2s, 256, 256, 128);

    // --- stage 0 features: x (N x 3) zero-padded to bf16 (N x 64) ---
    f32_to_bf16_pad<<<(N * 64 + 255) / 256, 256, 0, stream>>>(x, pbf, N, 3, 64);

    const bf16_t* wdS[3] = {wd1s, wd2s, wd2s};
    const bf16_t* weS[3] = {we1s, we2s, we2s};
    const float*  bdS[3] = {bd1, bd2, bd2};
    const float*  beS[3] = {be1, be2, be2};

    for (int stage = 0; stage < 3; ++stage) {
        row_sqnorm<<<(N + 255) / 256, 256, 0, stream>>>(pbf, sq, N);
        knn_wmma<<<N / 16, 256, 0, stream>>>(pbf, sq, knn, N);
        gemm_bf16_wmma<<<dim3(N / 16, 1), 128, 0, stream>>>(pbf, wdS[stage], bdS[stage], hd, N, 64, 64, 0);
        gemm_bf16_wmma<<<dim3(N / 16, 1), 128, 0, stream>>>(pbf, weS[stage], beS[stage], he, N, 64, 64, 0);
        gather_morpho<<<(N * 64 + 255) / 256, 256, 0, stream>>>(hd, he, knn, pbf, cat, 192, stage * 64, N);
    }

    // --- MLP head ---
    f32_to_bf16_pad<<<(N * 192 + 255) / 256, 256, 0, stream>>>(cat, catb, N, 192, 192);
    gemm_bf16_wmma<<<dim3(N / 16, 1024 / 64), 128, 0, stream>>>(catb, wl1s, blin1, h1, N, 192, 1024, 1);
    f32_to_bf16_pad<<<(N * 1024 + 255) / 256, 256, 0, stream>>>(h1, h1b, N, 1024, 1024);
    gemm_bf16_wmma<<<dim3(N / 16, 256 / 64), 128, 0, stream>>>(h1b, wm1s, bm1, h2, N, 1024, 256, 1);
    f32_to_bf16_pad<<<(N * 256 + 255) / 256, 256, 0, stream>>>(h2, h2b, N, 256, 256);
    gemm_bf16_wmma<<<dim3(N / 16, 128 / 64), 128, 0, stream>>>(h2b, wm2s, bm2, h3, N, 256, 128, 1);

    head_logsoftmax<<<N, 64, 0, stream>>>(h3, Wout, bout, (float*)d_out, N);
}
